// RNNModel_38182259261441
// MI455X (gfx1250) — compile-verified, matrix operand order checked by reading, not run on hardware
//
#include <hip/hip_runtime.h>
#include <hip/hip_bf16.h>

#define SEQ     256
#define BATCH   16
#define NLAYERS 2
#define NHID    1024
#define NTOKEN  50257

typedef __bf16 bf16_t;
typedef __attribute__((ext_vector_type(16))) __bf16        v16bf;
typedef __attribute__((ext_vector_type(8)))  float         v8f;
typedef __attribute__((ext_vector_type(4)))  unsigned int  u32x4;

// CDNA5 async global->LDS copy (ASYNCcnt path), signature confirmed by probe:
// (int4 AS1* src, int4 AS3* dst, imm offset, imm cpol)
#if defined(__gfx1250__) && __has_builtin(__builtin_amdgcn_global_load_async_to_lds_b128)
#define GEMM_ASYNC 1
typedef int v4i_vs __attribute__((vector_size(16)));
typedef __attribute__((address_space(1))) v4i_vs* async_gptr_t;
typedef __attribute__((address_space(3))) v4i_vs* async_lptr_t;
#define ASYNC_COPY(gp, lp) __builtin_amdgcn_global_load_async_to_lds_b128( \
        (async_gptr_t)(uintptr_t)(gp), (async_lptr_t)(void*)(lp), 0, 0)
#if __has_builtin(__builtin_amdgcn_s_wait_asynccnt)
#define WAIT_ASYNC() __builtin_amdgcn_s_wait_asynccnt(0)
#else
#define WAIT_ASYNC() asm volatile("s_wait_asynccnt 0" ::: "memory")
#endif
#else
#define GEMM_ASYNC 0
#define ASYNC_COPY(gp, lp) (*reinterpret_cast<u32x4*>(lp) = *reinterpret_cast<const u32x4*>(gp))
#define WAIT_ASYNC() ((void)0)
#endif

static __device__ __forceinline__ bf16_t f2bf(float f) {
    union { float f; unsigned u; } x; x.f = f;
    unsigned r = x.u + 0x7FFFu + ((x.u >> 16) & 1u);   // round-to-nearest-even
    unsigned short h = (unsigned short)(r >> 16);
    return __builtin_bit_cast(bf16_t, h);
}

union Frag { u32x4 q[2]; v16bf v; };

// ---------------------------------------------------------------------------
// Transpose + f32->bf16:  Bt[n][k] = bf16(B[k][n]).  32x32 tiles via LDS.
// ---------------------------------------------------------------------------
__global__ __launch_bounds__(256) void transpose_cvt_kernel(
    const float* __restrict__ B, bf16_t* __restrict__ Bt, int K, int N)
{
    __shared__ float tile[32][33];
    const int tx = threadIdx.x & 31;
    const int ty = threadIdx.x >> 5;            // 0..7
    const int kb = blockIdx.y * 32;
    const int nb = blockIdx.x * 32;
    #pragma unroll
    for (int i = 0; i < 32; i += 8) {
        int k = kb + ty + i, n = nb + tx;
        tile[ty + i][tx] = (k < K && n < N) ? B[(size_t)k * N + n] : 0.f;
    }
    __syncthreads();
    #pragma unroll
    for (int i = 0; i < 32; i += 8) {
        int n = nb + ty + i, k = kb + tx;
        if (n < N && k < K) Bt[(size_t)n * K + k] = f2bf(tile[tx][ty + i]);
    }
}

// ---------------------------------------------------------------------------
// GEMM:  C[M,N] = A[M,K](bf16) * Bt[N,K](bf16, pre-transposed) + bias
// Block 256 threads = 8 waves; tile (MT*16) x 128, K-step 32.
// Both operands stream to LDS with async b128 copies, double buffered
// (stage of tile k+1 overlaps WMMA of tile k).
// ---------------------------------------------------------------------------
template<int MT>
__global__ __launch_bounds__(256) void gemm_bfbf_kernel(
    const bf16_t* __restrict__ A, int lda,
    const bf16_t* __restrict__ Bt,
    const float*  __restrict__ bias,
    float* __restrict__ C, int ldc,
    int M, int N, int K)
{
    constexpr int BM  = MT * 16;
    constexpr int BN  = 128;
    constexpr int BK  = 32;
    constexpr int KP  = BK + 8;

    __shared__ bf16_t As[2][BM * KP];
    __shared__ bf16_t Bs[2][BN * KP];

    const int tid  = threadIdx.x;
    const int lane = tid & 31;
    const int wave = tid >> 5;
    const int m0   = blockIdx.y * BM;
    const int n0   = blockIdx.x * BN;
    const int r    = lane & 15;
    const int kb   = (lane >> 4) << 3;

    v8f zero = {0.f,0.f,0.f,0.f,0.f,0.f,0.f,0.f};
    v8f acc[MT];
    #pragma unroll
    for (int i = 0; i < MT; i++) acc[i] = zero;

    auto stage = [&](int kt, int sel) {
        bf16_t* as = &As[sel][0];
        bf16_t* bs = &Bs[sel][0];
        constexpr int NA = (BM * 4 + 255) / 256;
        #pragma unroll
        for (int it = 0; it < NA; ++it) {
            int idx = tid + it * 256;
            if ((BM * 4) % 256 == 0 || idx < BM * 4) {
                int row = idx >> 2, ch = (idx & 3) << 3;
                ASYNC_COPY(A + (size_t)(m0 + row) * lda + kt + ch, &as[row * KP + ch]);
            }
        }
        #pragma unroll
        for (int it = 0; it < 2; ++it) {           // BN*4 = 512 chunks
            int idx = tid + it * 256;
            int row = idx >> 2, ch = (idx & 3) << 3;
            int gr = n0 + row; if (gr > N - 1) gr = N - 1;   // clamp (edge cols discarded)
            ASYNC_COPY(Bt + (size_t)gr * K + kt + ch, &bs[row * KP + ch]);
        }
    };

    stage(0, 0);
    WAIT_ASYNC();
    __syncthreads();

    int sel = 0;
    for (int k0 = 0; k0 < K; k0 += BK) {
        if (k0 + BK < K) stage(k0 + BK, sel ^ 1);   // overlaps the WMMAs below

        Frag fb;
        const bf16_t* bp = &Bs[sel][(wave * 16 + r) * KP + kb];
        fb.q[0] = *reinterpret_cast<const u32x4*>(bp);
        fb.q[1] = *reinterpret_cast<const u32x4*>(bp + 16);

        #pragma unroll
        for (int g = 0; g < MT; g += 4) {
            Frag fa[4];
            #pragma unroll
            for (int u = 0; u < 4; u++) {
                if (g + u < MT) {
                    const bf16_t* ap = &As[sel][((g + u) * 16 + r) * KP + kb];
                    fa[u].q[0] = *reinterpret_cast<const u32x4*>(ap);
                    fa[u].q[1] = *reinterpret_cast<const u32x4*>(ap + 16);
                }
            }
            #pragma unroll
            for (int u = 0; u < 4; u++) {
                if (g + u < MT) {
                    acc[g + u] = __builtin_amdgcn_wmma_f32_16x16x32_bf16(
                        false, fa[u].v, false, fb.v, (short)0, acc[g + u], false, false);
                }
            }
        }

        WAIT_ASYNC();
        __syncthreads();
        sel ^= 1;
    }

    const int n = n0 + wave * 16 + r;
    if (n < N) {
        float bv = bias ? bias[n] : 0.f;
        const int mhi = (lane >> 4) ? 8 : 0;
        #pragma unroll
        for (int mt = 0; mt < MT; mt++) {
            #pragma unroll
            for (int i = 0; i < 8; i++) {
                C[(size_t)(m0 + mt * 16 + mhi + i) * ldc + n] = acc[mt][i] + bv;
            }
        }
    }
}

// ---------------------------------------------------------------------------
// Fused LSTM layer scan: one launch runs all SEQ timesteps.
// Grid = 32 blocks. Block bn owns the four gate strips {g*H + bn*32 ..+32},
// so the cell update is block-local; c lives in registers for the whole scan.
// This block's 128 Wh rows (264KB bf16) are cached in LDS ONCE and reused by
// all 256 steps -> steady-state k-loop is pure ds_load_b128 + v_wmma with no
// barriers and no global traffic. h_t broadcasts through a ping-pong global
// buffer with one device-wide atomic phase barrier per step.
// ---------------------------------------------------------------------------
__global__ __launch_bounds__(256) void lstm_layer_kernel(
    const bf16_t* __restrict__ Wht,   // (4H, H) bf16 (transposed Wh)
    const float*  __restrict__ xw,    // (S, B, 4H) = x@Wi + bi
    const float*  __restrict__ bh,    // (4H)
    const float*  __restrict__ c0,    // (B, H) initial c
    bf16_t* __restrict__ hb,          // (2, B, H) h ping-pong broadcast
    bf16_t* __restrict__ ybf,         // (S, B, H) layer output bf16
    float*  __restrict__ yf32,        // (S, B, H) or nullptr (final x)
    float*  __restrict__ h_out,       // final h (B, H)
    float*  __restrict__ c_out,       // final c (B, H)
    unsigned* __restrict__ ctr,       // global phase counter
    int phase_base)
{
    constexpr int WS = NHID + 8;             // padded K stride (halves)
    constexpr int PS = 132;
    __shared__ bf16_t Wc[128 * WS];          // weight cache: 264KB, loaded once
    __shared__ bf16_t Hs[BATCH * WS];        // full h_t: 33KB
    __shared__ float  Ps[BATCH * PS];        // pre-activations 16 x 128

    const int tid  = threadIdx.x;
    const int lane = tid & 31;
    const int wave = tid >> 5;
    const int bn   = blockIdx.x;             // 0..31
    const int r    = lane & 15;
    const int kb   = (lane >> 4) << 3;

    v8f zero = {0.f,0.f,0.f,0.f,0.f,0.f,0.f,0.f};

    // ---- cache this block's 128 gate-strip weight rows in LDS (once) ----
    #pragma unroll 8
    for (int it = 0; it < 64; ++it) {
        int idx  = tid + it * 256;           // 0..16383 chunks of 8 halves
        int row  = idx >> 7;                 // 0..127
        int ch   = (idx & 127) << 3;         // 0..1016
        int grow = (row >> 5) * NHID + bn * 32 + (row & 31);
        ASYNC_COPY(Wht + (size_t)grow * NHID + ch, &Wc[row * WS + ch]);
    }

    // per-thread cell state: 2 elements, resident in registers for all steps
    float creg[2];
    int   eb[2], ej[2];
    #pragma unroll
    for (int p = 0; p < 2; p++) {
        int e  = tid + p * 256;              // 0..511
        eb[p]  = e >> 5;                     // batch
        ej[p]  = e & 31;                     // j within this block's strip
        creg[p] = c0[eb[p] * NHID + bn * 32 + ej[p]];
    }

    WAIT_ASYNC();
    __syncthreads();

    for (int t = 0; t < SEQ; t++) {
        // ---- load full h_t into LDS (16 x 1024 bf16 = 32KB) ----
        const bf16_t* hcur = hb + (size_t)(t & 1) * BATCH * NHID;
        #pragma unroll
        for (int it = 0; it < 8; ++it) {
            int idx = tid + it * 256;
            int row = idx >> 7;
            int ch  = (idx & 127) << 3;
            ASYNC_COPY(hcur + (size_t)row * NHID + ch, &Hs[row * WS + ch]);
        }
        WAIT_ASYNC();
        __syncthreads();

        // ---- pre = h @ Wh (this block's 128 cols): pure LDS + WMMA ----
        v8f acc = zero;
        #pragma unroll 4
        for (int k0 = 0; k0 < NHID; k0 += 32) {
            Frag fa, fb;
            const bf16_t* ap = &Hs[r * WS + k0 + kb];
            fa.q[0] = *reinterpret_cast<const u32x4*>(ap);
            fa.q[1] = *reinterpret_cast<const u32x4*>(ap + 16);
            const bf16_t* bp = &Wc[(wave * 16 + r) * WS + k0 + kb];
            fb.q[0] = *reinterpret_cast<const u32x4*>(bp);
            fb.q[1] = *reinterpret_cast<const u32x4*>(bp + 16);
            acc = __builtin_amdgcn_wmma_f32_16x16x32_bf16(
                false, fa.v, false, fb.v, (short)0, acc, false, false);
        }

        // ---- dump pre-activations to LDS (block cols 0..127 = gate*32+jj) ----
        {
            int bc  = wave * 16 + r;
            int mhi = (lane >> 4) ? 8 : 0;
            #pragma unroll
            for (int i = 0; i < 8; i++) Ps[(mhi + i) * PS + bc] = acc[i];
        }
        __syncthreads();

        // ---- cell update: 2 elements per thread ----
        bf16_t* hnext = hb + (size_t)((t + 1) & 1) * BATCH * NHID;
        #pragma unroll
        for (int p = 0; p < 2; p++) {
            int b  = eb[p], jj = ej[p];
            int jg = bn * 32 + jj;
            const float* xwt = xw + ((size_t)t * BATCH + b) * (4 * NHID);
            float pf = Ps[b * PS + jj]      + xwt[jg]            + bh[jg];
            float pi = Ps[b * PS + 32 + jj] + xwt[NHID + jg]     + bh[NHID + jg];
            float po = Ps[b * PS + 64 + jj] + xwt[2 * NHID + jg] + bh[2 * NHID + jg];
            float pg = Ps[b * PS + 96 + jj] + xwt[3 * NHID + jg] + bh[3 * NHID + jg];

            float fg = 1.f / (1.f + __expf(-pf));
            float ig = 1.f / (1.f + __expf(-pi));
            float og = 1.f / (1.f + __expf(-po));
            float gg = tanhf(pg);

            float cn = fg * creg[p] + ig * gg;
            float hn = og * tanhf(cn);
            creg[p] = cn;

            size_t oidx = (size_t)b * NHID + jg;
            hnext[oidx] = f2bf(hn);
            ybf[(size_t)t * BATCH * NHID + oidx] = f2bf(hn);
            if (yf32) yf32[(size_t)t * BATCH * NHID + oidx] = hn;
            if (t == SEQ - 1) { h_out[oidx] = hn; c_out[oidx] = cn; }
        }

        // ---- device-wide phase barrier: h_{t+1} visible before next step ----
        __threadfence();
        __syncthreads();
        if (tid == 0) {
            __hip_atomic_fetch_add(ctr, 1u, __ATOMIC_ACQ_REL, __HIP_MEMORY_SCOPE_AGENT);
            unsigned target = 32u * (unsigned)(phase_base + t + 1);
            while (__hip_atomic_load(ctr, __ATOMIC_ACQUIRE, __HIP_MEMORY_SCOPE_AGENT) < target) {
                __builtin_amdgcn_s_sleep(1);
            }
        }
        __syncthreads();
    }
}

// ---------------------------------------------------------------------------
__global__ __launch_bounds__(256) void embed_kernel(
    const int* __restrict__ tokens, const float* __restrict__ emb,
    bf16_t* __restrict__ xbf)
{
    int idx = blockIdx.x * 256 + threadIdx.x;
    int sb  = idx >> 10;
    int j   = idx & 1023;
    int tok = tokens[sb];
    xbf[idx] = f2bf(emb[(size_t)tok * NHID + j]);
}

__global__ __launch_bounds__(256) void init_state_kernel(
    const float* __restrict__ h0, bf16_t* __restrict__ hb0)
{
    int idx = blockIdx.x * 256 + threadIdx.x;      // < BATCH*NHID
    hb0[idx] = f2bf(h0[idx]);
}

__global__ void zero_ctr_kernel(unsigned* ctr) { *ctr = 0u; }

// ---------------------------------------------------------------------------
extern "C" void kernel_launch(void* const* d_in, const int* in_sizes, int n_in,
                              void* d_out, int out_size, void* d_ws, size_t ws_size,
                              hipStream_t stream)
{
    const int*   tokens = (const int*)  d_in[0];
    const float* hidden = (const float*)d_in[1];
    const float* emb    = (const float*)d_in[2];
    const float* wi     = (const float*)d_in[3];
    const float* bi     = (const float*)d_in[4];
    const float* wh     = (const float*)d_in[5];
    const float* bh     = (const float*)d_in[6];
    const float* wdec   = (const float*)d_in[7];
    const float* bdec   = (const float*)d_in[8];

    float* out_logits = (float*)d_out;                                   // (S,B,NTOKEN)
    float* out_hidden = out_logits + (size_t)SEQ * BATCH * NTOKEN;       // (L,2,B,H)
    float* out_x      = out_hidden + (size_t)NLAYERS * 2 * BATCH * NHID; // (S,B,H)

    char* w = (char*)d_ws;
    unsigned* ctr  = (unsigned*)w;  w += 256;
    bf16_t* xbf0   = (bf16_t*)w;    w += (size_t)SEQ * BATCH * NHID * 2;
    bf16_t* xbf1   = (bf16_t*)w;    w += (size_t)SEQ * BATCH * NHID * 2;
    float*  xw     = (float*) w;    w += (size_t)SEQ * BATCH * 4 * NHID * 4;
    bf16_t* hb     = (bf16_t*)w;    w += (size_t)2 * BATCH * NHID * 2;   // ping-pong h
    bf16_t* wi_t   = (bf16_t*)w;    w += (size_t)4 * NHID * NHID * 2;
    bf16_t* wh_t   = (bf16_t*)w;    w += (size_t)4 * NHID * NHID * 2;
    bf16_t* wdec_t = (bf16_t*)w;    w += (size_t)NTOKEN * NHID * 2;

    const int SB = SEQ * BATCH;                  // 4096
    const int BHblocks = (BATCH * NHID) / 256;   // 64

    zero_ctr_kernel<<<1, 1, 0, stream>>>(ctr);
    embed_kernel<<<(size_t)SB * NHID / 256, 256, 0, stream>>>(tokens, emb, xbf0);

    bf16_t* xin  = xbf0;
    bf16_t* xout = xbf1;

    for (int l = 0; l < NLAYERS; l++) {
        const float* wi_l = wi + (size_t)l * NHID * 4 * NHID;
        const float* wh_l = wh + (size_t)l * NHID * 4 * NHID;
        const float* h0   = hidden + (size_t)l * 2 * BATCH * NHID;
        const float* c0   = h0 + (size_t)BATCH * NHID;
        float* hslot = out_hidden + (size_t)l * 2 * BATCH * NHID;
        float* cslot = hslot + (size_t)BATCH * NHID;

        // transposed bf16 weights (one-time per layer)
        transpose_cvt_kernel<<<dim3(4 * NHID / 32, NHID / 32), 256, 0, stream>>>(
            wi_l, wi_t, NHID, 4 * NHID);
        transpose_cvt_kernel<<<dim3(4 * NHID / 32, NHID / 32), 256, 0, stream>>>(
            wh_l, wh_t, NHID, 4 * NHID);

        // xw = xin @ Wi + bi   (4096 x 1024) @ (1024 x 4096)
        gemm_bfbf_kernel<8><<<dim3(4 * NHID / 128, SB / 128), 256, 0, stream>>>(
            xin, NHID, wi_t, bi + (size_t)l * 4 * NHID, xw, 4 * NHID,
            SB, 4 * NHID, NHID);

        // h0 -> broadcast buffer 0
        init_state_kernel<<<BHblocks, 256, 0, stream>>>(h0, hb);

        // full sequential scan in ONE kernel
        lstm_layer_kernel<<<32, 256, 0, stream>>>(
            wh_t, xw, bh + (size_t)l * 4 * NHID, c0, hb,
            xout,
            (l == NLAYERS - 1) ? out_x : nullptr,
            hslot, cslot, ctr, l * SEQ);

        bf16_t* tmp = xin; xin = xout; xout = tmp;
    }

    // decoder weights: transpose+convert once, then pure-async GEMM
    transpose_cvt_kernel<<<dim3((NTOKEN + 31) / 32, NHID / 32), 256, 0, stream>>>(
        wdec, wdec_t, NHID, NTOKEN);

    gemm_bfbf_kernel<8><<<dim3((NTOKEN + 127) / 128, SB / 128), 256, 0, stream>>>(
        xin, NHID, wdec_t, bdec, out_logits, NTOKEN,
        SB, NTOKEN, NHID);
}
